// OctreeResBlock_24146306138641
// MI455X (gfx1250) — compile-verified, matrix operand order checked by reading, not run on hardware
//
#include <hip/hip_runtime.h>

typedef __attribute__((ext_vector_type(16))) __bf16 v16bf;
typedef __attribute__((ext_vector_type(8)))  __bf16 v8bf;
typedef __attribute__((ext_vector_type(8)))  float  v8f;
typedef __attribute__((ext_vector_type(4)))  int    v4i;

#define NPTS 60000
#define CIN  256
#define CB   128
#define COUT 512
#define KNB  27

#if defined(__has_builtin)
#if __has_builtin(__builtin_amdgcn_global_load_async_to_lds_b128)
#define USE_ASYNC_BUILTIN 1
#endif
#if __has_builtin(__builtin_amdgcn_s_wait_asynccnt)
#define USE_WAIT_BUILTIN 1
#endif
#endif
#ifndef USE_ASYNC_BUILTIN
#define USE_ASYNC_BUILTIN 0
#endif
#ifndef USE_WAIT_BUILTIN
#define USE_WAIT_BUILTIN 0
#endif

typedef __attribute__((address_space(1))) v4i glb_v4i;  // global int4
typedef __attribute__((address_space(3))) v4i lds_v4i;  // LDS int4

__device__ __forceinline__ v8f wmma_bf16(v16bf a, v16bf b, v8f c) {
  return __builtin_amdgcn_wmma_f32_16x16x32_bf16(false, a, false, b, (short)0, c,
                                                 false, false);
}

__device__ __forceinline__ v16bf cat16(v8bf lo, v8bf hi) {
  return __builtin_shufflevector(lo, hi, 0, 1, 2, 3, 4, 5, 6, 7, 8, 9, 10, 11, 12,
                                 13, 14, 15);
}

// CDNA5 async direct-to-LDS 16B copy (no VGPR round trip, tracked by ASYNCcnt).
// Builtin path carries proper memory semantics on the addrspace(3) destination
// so the later ds_loads of the staged data stay live. Asm fallback escapes the
// LDS address via ptrtoint of the addrspace(3) pointer for the same reason.
__device__ __forceinline__ void async_copy16(__bf16* ldst, const __bf16* gsrc) {
#if USE_ASYNC_BUILTIN
  __builtin_amdgcn_global_load_async_to_lds_b128((glb_v4i*)gsrc, (lds_v4i*)ldst, 0,
                                                 0);
#else
  unsigned loff =
      (unsigned)(size_t)(__attribute__((address_space(3))) __bf16*)ldst;
  asm volatile("global_load_async_to_lds_b128 %0, %1, off" ::"v"(loff), "v"(gsrc)
               : "memory");
#endif
}

__device__ __forceinline__ void wait_async0() {
#if USE_WAIT_BUILTIN
  __builtin_amdgcn_s_wait_asynccnt(0);
#else
  asm volatile("s_wait_asynccnt 0x0" ::: "memory");
#endif
}

// ---------------- prep kernels ----------------

__global__ void cvt_bf16(const float* __restrict__ s, __bf16* __restrict__ d, int n) {
  int i = blockIdx.x * 256 + threadIdx.x;
  if (i < n) d[i] = (__bf16)s[i];
}

// W3 [k][c][d] fp32 -> W3t [k][d][c] bf16 (so B-fragments are K-contiguous)
__global__ void w3_transpose(const float* __restrict__ s, __bf16* __restrict__ d) {
  int i = blockIdx.x * 256 + threadIdx.x;
  if (i < KNB * CB * CB) {
    int k = i >> 14, rem = i & 16383;
    int c = rem >> 7, dd = rem & 127;
    d[(k << 14) + (dd << 7) + c] = (__bf16)s[i];
  }
}

// x [CIN, NPTS] fp32 -> hb [NPTS, CIN] bf16, LDS-tiled transpose
__global__ __launch_bounds__(256) void transpose_x(const float* __restrict__ x,
                                                   __bf16* __restrict__ hb) {
  __shared__ float t[32][33];
  int tx = threadIdx.x, ty = threadIdx.y;  // 32 x 8
  int nb = blockIdx.x * 32, cb = blockIdx.y * 32;
#pragma unroll
  for (int i = 0; i < 4; ++i)
    t[ty + 8 * i][tx] = x[(size_t)(cb + ty + 8 * i) * NPTS + nb + tx];
  __syncthreads();
#pragma unroll
  for (int i = 0; i < 4; ++i)
    hb[(size_t)(nb + ty + 8 * i) * CIN + cb + tx] = (__bf16)t[tx][ty + 8 * i];
}

// ---------------- WMMA GEMM: Y[N,CO] = A[N,KD](bf16) @ W[CO,KD]^T ----------------
// Workgroup: 8 waves; wave handles 16 rows x 128 cols; W slice cached in LDS
// via async direct-to-LDS copies.

template <int CO, int KD>
__global__ __launch_bounds__(256) void gemm_xwt(const __bf16* __restrict__ A,
                                                const __bf16* __restrict__ W,
                                                float* __restrict__ Y, int nTiles) {
  __shared__ __bf16 sW[128 * KD];
  int tid = threadIdx.x;
  // stage weight slice (128 output cols x KD) into LDS, bypassing VGPRs
  const __bf16* wsrc = W + (size_t)blockIdx.y * 128 * KD;
  constexpr int NV = 128 * KD / 8;  // 16B vectors
#pragma unroll
  for (int v = tid; v < NV; v += 256) async_copy16(sW + v * 8, wsrc + v * 8);
  wait_async0();
  __syncthreads();

  int wave = tid >> 5, lane = tid & 31;
  int tile = blockIdx.x * 8 + wave;
  if (tile >= nTiles) return;

  int r = lane & 15, kb = (lane >> 4) * 8;
  const __bf16* Arow = A + (size_t)(tile * 16 + r) * KD;
  v8f acc[8] = {};
#pragma unroll
  for (int kc = 0; kc < KD / 32; ++kc) {
    int base = kc * 32 + kb;
    v16bf af = cat16(*(const v8bf*)(Arow + base), *(const v8bf*)(Arow + base + 16));
    v16bf bfr[8];
#pragma unroll
    for (int j = 0; j < 8; ++j) {
      const __bf16* Brow = sW + (size_t)(j * 16 + r) * KD + base;
      bfr[j] = cat16(*(const v8bf*)Brow, *(const v8bf*)(Brow + 16));
    }
#pragma unroll
    for (int j = 0; j < 8; ++j) acc[j] = wmma_bf16(af, bfr[j], acc[j]);
  }
  int rowhi = 8 * (lane >> 4), col0 = blockIdx.y * 128 + (lane & 15);
#pragma unroll
  for (int j = 0; j < 8; ++j)
#pragma unroll
    for (int v = 0; v < 8; ++v)
      Y[(size_t)(tile * 16 + v + rowhi) * CO + col0 + j * 16] = acc[j][v];
}

// ---------------- octree conv: y2[n,d] = sum_k sum_c c1b[neigh[n,k],c] W3t[k][d][c]
// c1b has a zeroed row at index NPTS, so empty neighbors (idx==NPTS) need no
// per-lane select. Weight slices are double-buffered: the async DMA of slice
// k+1 overlaps the 32 WMMAs of slice k; one barrier per iteration.
__global__ __launch_bounds__(256) void octconv(const __bf16* __restrict__ c1b,
                                               const int* __restrict__ neigh,
                                               const __bf16* __restrict__ w3t,
                                               float* __restrict__ y2) {
  __shared__ __bf16 sW[2 * 128 * 128];  // 2 x 32KB
  int tid = threadIdx.x, wave = tid >> 5, lane = tid & 31;
  int tileBase = blockIdx.x * 128 + wave * 16;
  int r = lane & 15, kb = (lane >> 4) * 8;
  int node = tileBase + r;
  bool inr = node < NPTS;
  v8f acc[8] = {};

  // preload slice 0 into buffer 0
#pragma unroll
  for (int v = 0; v < 8; ++v)
    async_copy16(sW + (tid + v * 256) * 8, w3t + (size_t)(tid + v * 256) * 8);
  wait_async0();
  __syncthreads();

  for (int k = 0; k < KNB; ++k) {
    if (k + 1 < KNB) {  // kick off DMA of next slice into the alternate buffer
      __bf16* nxt = sW + (size_t)((k + 1) & 1) * 16384;
      const __bf16* src = w3t + (size_t)(k + 1) * 16384;
#pragma unroll
      for (int v = 0; v < 8; ++v)
        async_copy16(nxt + (tid + v * 256) * 8, src + (size_t)(tid + v * 256) * 8);
    }
    const __bf16* sWk = sW + (size_t)(k & 1) * 16384;
    int idx = inr ? neigh[node * KNB + k] : NPTS;  // NPTS -> zero row
    const __bf16* g = c1b + (size_t)idx * CB;
#pragma unroll
    for (int cc = 0; cc < 4; ++cc) {
      int base = cc * 32 + kb;
      v16bf af = cat16(*(const v8bf*)(g + base), *(const v8bf*)(g + base + 16));
      v16bf bfr[8];
#pragma unroll
      for (int j = 0; j < 8; ++j) {
        const __bf16* Brow = sWk + (size_t)(j * 16 + r) * CB + base;
        bfr[j] = cat16(*(const v8bf*)Brow, *(const v8bf*)(Brow + 16));
      }
#pragma unroll
      for (int j = 0; j < 8; ++j) acc[j] = wmma_bf16(af, bfr[j], acc[j]);
    }
    wait_async0();   // our portion of slice k+1 landed
    __syncthreads(); // everyone's portion landed / done reading slice k
  }
  int rowhi = 8 * (lane >> 4), col0 = lane & 15;
#pragma unroll
  for (int j = 0; j < 8; ++j)
#pragma unroll
    for (int v = 0; v < 8; ++v) {
      int nr = tileBase + v + rowhi;
      if (nr < NPTS) y2[(size_t)nr * CB + col0 + j * 16] = acc[j][v];
    }
}

// ---------------- BN statistics / params / epilogues ----------------

template <int C>
__global__ __launch_bounds__(256) void reduce_stats(const float* __restrict__ y,
                                                    float* __restrict__ st) {
  constexpr int CPT = (C > 256) ? (C / 256) : 1;  // channels per thread
  constexpr int TPR = (C <= 256) ? (256 / C) : 1; // row-parallel threads
  int tid = threadIdx.x;
  int c = (C <= 256) ? (tid & (C - 1)) : tid;
  int r0 = blockIdx.x * 512 + ((C <= 256) ? (tid / C) : 0);
  int rend = min(blockIdx.x * 512 + 512, NPTS);
  float s[CPT] = {}, q[CPT] = {};
  for (int rr = r0; rr < rend; rr += TPR) {
#pragma unroll
    for (int u = 0; u < CPT; ++u) {
      float v = y[(size_t)rr * C + c + u * 256];
      s[u] += v;
      q[u] += v * v;
    }
  }
#pragma unroll
  for (int u = 0; u < CPT; ++u) {
    atomicAdd(&st[c + u * 256], s[u]);
    atomicAdd(&st[C + c + u * 256], q[u]);
  }
}

__global__ void bn_params(const float* __restrict__ st, const float* __restrict__ g,
                          const float* __restrict__ b, float* __restrict__ prm, int C) {
  int c = threadIdx.x;
  if (c < C) {
    float mean = st[c] * (1.0f / NPTS);
    float var = st[C + c] * (1.0f / NPTS) - mean * mean;  // biased var, matches ref
    float sc = g[c] * rsqrtf(var + 1e-5f);
    prm[c] = sc;
    prm[C + c] = b[c] - mean * sc;
  }
}

template <int C>
__global__ void apply_relu_bf16(const float* __restrict__ y,
                                const float* __restrict__ prm,
                                __bf16* __restrict__ out) {
  int i = blockIdx.x * 256 + threadIdx.x;
  if (i < NPTS * C) {
    int c = i & (C - 1);
    float v = fmaf(y[i], prm[c], prm[C + c]);
    out[i] = (__bf16)fmaxf(v, 0.0f);
  }
}

// out[c*N + n] = relu(bn(y3) + bn(ysc)); LDS-tiled so reads AND the 123MB of
// stores are coalesced (this kernel is pure HBM bandwidth).
__global__ __launch_bounds__(256) void final_combine(const float* __restrict__ y3,
                                                     const float* __restrict__ ysc,
                                                     const float* __restrict__ p3,
                                                     const float* __restrict__ pc,
                                                     float* __restrict__ out) {
  __shared__ float t[32][33];
  int tx = threadIdx.x, ty = threadIdx.y;  // 32 x 8
  int nb = blockIdx.x * 32, cb = blockIdx.y * 32;
#pragma unroll
  for (int i = 0; i < 4; ++i) {
    int n = nb + ty + 8 * i, c = cb + tx;
    size_t ix = (size_t)n * COUT + c;
    float a = fmaf(y3[ix], p3[c], p3[COUT + c]);
    float s = fmaf(ysc[ix], pc[c], pc[COUT + c]);
    t[ty + 8 * i][tx] = fmaxf(a + s, 0.0f);  // t[n_local][c_local]
  }
  __syncthreads();
#pragma unroll
  for (int i = 0; i < 4; ++i) {
    int c = cb + ty + 8 * i, n = nb + tx;
    out[(size_t)c * NPTS + n] = t[tx][ty + 8 * i];
  }
}

// ---------------- launch ----------------

extern "C" void kernel_launch(void* const* d_in, const int* in_sizes, int n_in,
                              void* d_out, int out_size, void* d_ws, size_t ws_size,
                              hipStream_t stream) {
  const float* x    = (const float*)d_in[0];
  const int*   nb   = (const int*)d_in[1];
  const float* W1a  = (const float*)d_in[2];
  const float* g1a  = (const float*)d_in[3];
  const float* b1a  = (const float*)d_in[4];
  const float* W3   = (const float*)d_in[5];
  const float* g3   = (const float*)d_in[6];
  const float* b3   = (const float*)d_in[7];
  const float* W1b  = (const float*)d_in[8];
  const float* g1b  = (const float*)d_in[9];
  const float* b1b  = (const float*)d_in[10];
  const float* Wc   = (const float*)d_in[11];
  const float* gc   = (const float*)d_in[12];
  const float* bc   = (const float*)d_in[13];
  float* out = (float*)d_out;

  char* w = (char*)d_ws;
  size_t off = 0;
  auto alloc = [&](size_t bytes) {
    void* p = w + off;
    off = (off + bytes + 255) & ~(size_t)255;
    return p;
  };
  __bf16* hb   = (__bf16*)alloc((size_t)NPTS * CIN * 2);
  __bf16* w1ab = (__bf16*)alloc((size_t)CB * CIN * 2);
  __bf16* w3t  = (__bf16*)alloc((size_t)KNB * CB * CB * 2);
  __bf16* w1bb = (__bf16*)alloc((size_t)COUT * CB * 2);
  __bf16* wcb  = (__bf16*)alloc((size_t)COUT * CIN * 2);
  float* y1  = (float*)alloc((size_t)NPTS * CB * 4);
  __bf16* c1b = (__bf16*)alloc((size_t)(NPTS + 1) * CB * 2);  // +1 zero row
  float* y2  = (float*)alloc((size_t)NPTS * CB * 4);
  __bf16* c2b = (__bf16*)alloc((size_t)NPTS * CB * 2);
  float* y3  = (float*)alloc((size_t)NPTS * COUT * 4);
  float* ysc = (float*)alloc((size_t)NPTS * COUT * 4);
  float* st1 = (float*)alloc(2 * CB * 4);
  float* st2 = (float*)alloc(2 * CB * 4);
  float* st3 = (float*)alloc(2 * COUT * 4);
  float* stc = (float*)alloc(2 * COUT * 4);
  float* p1 = (float*)alloc(2 * CB * 4);
  float* p2 = (float*)alloc(2 * CB * 4);
  float* p3 = (float*)alloc(2 * COUT * 4);
  float* pc = (float*)alloc(2 * COUT * 4);

  const int nTiles = NPTS / 16;               // 3750
  const int gB = (nTiles + 7) / 8;            // 469 blocks of 8 waves

  // zero BN stat accumulators + the empty-neighbor row of c1b
  (void)hipMemsetAsync(st1, 0, (size_t)((char*)p1 - (char*)st1), stream);
  (void)hipMemsetAsync(c1b + (size_t)NPTS * CB, 0, CB * 2, stream);

  // weight prep
  cvt_bf16<<<(CB * CIN + 255) / 256, 256, 0, stream>>>(W1a, w1ab, CB * CIN);
  cvt_bf16<<<(COUT * CB + 255) / 256, 256, 0, stream>>>(W1b, w1bb, COUT * CB);
  cvt_bf16<<<(COUT * CIN + 255) / 256, 256, 0, stream>>>(Wc, wcb, COUT * CIN);
  w3_transpose<<<(KNB * CB * CB + 255) / 256, 256, 0, stream>>>(W3, w3t);
  transpose_x<<<dim3(NPTS / 32, CIN / 32), dim3(32, 8), 0, stream>>>(x, hb);

  // stage 1: 1x1 conv + BN + ReLU
  gemm_xwt<CB, CIN><<<dim3(gB, 1), 256, 0, stream>>>(hb, w1ab, y1, nTiles);
  reduce_stats<CB><<<(NPTS + 511) / 512, 256, 0, stream>>>(y1, st1);
  bn_params<<<1, CB, 0, stream>>>(st1, g1a, b1a, p1, CB);
  apply_relu_bf16<CB><<<(NPTS * CB + 255) / 256, 256, 0, stream>>>(y1, p1, c1b);

  // stage 2: sparse octree 3x3 conv + BN + ReLU
  octconv<<<(NPTS + 127) / 128, 256, 0, stream>>>(c1b, nb, w3t, y2);
  reduce_stats<CB><<<(NPTS + 511) / 512, 256, 0, stream>>>(y2, st2);
  bn_params<<<1, CB, 0, stream>>>(st2, g3, b3, p2, CB);
  apply_relu_bf16<CB><<<(NPTS * CB + 255) / 256, 256, 0, stream>>>(y2, p2, c2b);

  // stage 3 + shortcut
  gemm_xwt<COUT, CB><<<dim3(gB, 4), 256, 0, stream>>>(c2b, w1bb, y3, nTiles);
  gemm_xwt<COUT, CIN><<<dim3(gB, 4), 256, 0, stream>>>(hb, wcb, ysc, nTiles);
  reduce_stats<COUT><<<(NPTS + 511) / 512, 256, 0, stream>>>(y3, st3);
  reduce_stats<COUT><<<(NPTS + 511) / 512, 256, 0, stream>>>(ysc, stc);
  bn_params<<<1, COUT, 0, stream>>>(st3, g1b, b1b, p3, COUT);
  bn_params<<<1, COUT, 0, stream>>>(stc, gc, bc, pc, COUT);

  // fused residual add + ReLU + transpose to [Cout, N]
  final_combine<<<dim3(NPTS / 32, COUT / 32), dim3(32, 8), 0, stream>>>(y3, ysc, p3,
                                                                        pc, out);
}
